// KOBE_34462817583803
// MI455X (gfx1250) — matrix operand order repaired and checked by reading, not compiled, not run on hardware
//
#include <hip/hip_runtime.h>
#include <hip/hip_bf16.h>

typedef __attribute__((ext_vector_type(16))) _Float16 v16h;
typedef __attribute__((ext_vector_type(8)))  float    v8f;

#define ORDER_    12
#define NUMBITS_  32
#define T_        4095   // 2^12 - 1
#define B_        4096
#define N64       64     // WHT factor size: H_4096 = H_64 (x) H_64

// ---------------------------------------------------------------------------
// Kernel 1: one workgroup (512 threads = 16 wave32) does
//   (a) scatter theta[t] -> Theta[mask(t)] (f16, LDS), mask from idx_pad
//   (b) G = H64 * Theta        (WMMA f16->f32, H synthesized via popcount)
//   (c) F = G * H64            (WMMA, G split into hi/lo f16 for precision)
//   (d) F (4096 f32) -> workspace
// ---------------------------------------------------------------------------
__global__ __launch_bounds__(512)
void KOBE_wht_kernel(const float* __restrict__ theta,
                     const int* __restrict__ idx_pad,
                     float* __restrict__ ws_F) {
    __shared__ _Float16 sTheta[N64 * N64];  // 8 KB  : Theta in mask order
    __shared__ float    sG[N64 * N64];      // 16 KB : G = H * Theta (exact f32)

    const int tid  = threadIdx.x;
    const int lane = tid & 31;
    const int wave = tid >> 5;          // 0..15
    const int ti   = wave >> 2;         // tile row   (0..3)
    const int tj   = wave & 3;          // tile col   (0..3)

    // ---- (a) scatter theta into Walsh-mask order --------------------------
    if (tid == 0) sTheta[0] = (_Float16)0.0f;   // empty subset: weight 0
    for (int t = tid; t < T_; t += 512) {
        int mask = 0;
        #pragma unroll
        for (int k = 0; k < ORDER_; ++k) {
            int e = idx_pad[t * ORDER_ + k];
            if (e < ORDER_) mask |= (1 << e);   // pad entries point at col 32
        }
        sTheta[mask] = (_Float16)theta[t];      // masks are a permutation of 1..4095
    }
    __syncthreads();

    // Per-lane fragment geometry (contiguous k order chosen by this code, wave32):
    //  A (16x32 f16): M = lane%16; halves j: K = kbaseA + j,
    //                 kbaseA = 16 for lanes 16..31.
    //  B (32x16 f16): N = lane%16; halves j: K = kbaseB + j, kbaseB = 16 for lanes 16..31.
    //  C/D (16x16 f32): element r: M = r + (lane>=16 ? 8 : 0), N = lane%16.
    const int mloc   = lane & 15;
    const int kbaseA = (lane >> 4) ? 16 : 0;
    const int kbaseB = (lane >> 4) ? 16 : 0;
    const int hi8    = (lane >> 4) ? 8 : 0;

    // ---- (b) G = H * Theta : G[x][c] = sum_m (-1)^pc(x&m) Theta[m][c] -----
    v8f acc = {};
    #pragma unroll
    for (int ks = 0; ks < 2; ++ks) {
        v16h a, b;
        #pragma unroll
        for (int j = 0; j < 16; ++j) {
            int kA = ks * 32 + kbaseA + j;   // m index
            int x  = ti * 16 + mloc;                           // output row
            a[j] = (__popc(x & kA) & 1) ? (_Float16)(-1.0f) : (_Float16)(1.0f);
            int kB = ks * 32 + kbaseB + j;                     // m index
            b[j] = sTheta[kB * N64 + tj * 16 + mloc];          // Theta[m][c]
        }
        acc = __builtin_amdgcn_wmma_f32_16x16x32_f16(
                  false, a, false, b, (short)0, acc, false, false);
    }
    #pragma unroll
    for (int r = 0; r < 8; ++r)
        sG[(ti * 16 + r + hi8) * N64 + tj * 16 + mloc] = acc[r];
    __syncthreads();

    // ---- (c) F = G * H : F[x][y] = sum_c G[x][c] (-1)^pc(c&y) -------------
    v8f f = {};
    #pragma unroll
    for (int ks = 0; ks < 2; ++ks) {
        v16h ahi, alo, bh;
        #pragma unroll
        for (int j = 0; j < 16; ++j) {
            int kA = ks * 32 + kbaseA + j;   // c index
            float g = sG[(ti * 16 + mloc) * N64 + kA];
            _Float16 h = (_Float16)g;                          // hi part
            ahi[j] = h;
            alo[j] = (_Float16)(g - (float)h);                 // compensation
            int kB = ks * 32 + kbaseB + j;                     // c index
            int y  = tj * 16 + mloc;                           // output col
            bh[j] = (__popc(kB & y) & 1) ? (_Float16)(-1.0f) : (_Float16)(1.0f);
        }
        f = __builtin_amdgcn_wmma_f32_16x16x32_f16(
                false, ahi, false, bh, (short)0, f, false, false);
        f = __builtin_amdgcn_wmma_f32_16x16x32_f16(
                false, alo, false, bh, (short)0, f, false, false);
    }

    // ---- (d) F tile -> workspace (F[xh*64+xl] = WHT(theta')[x]) -----------
    #pragma unroll
    for (int r = 0; r < 8; ++r)
        ws_F[(ti * 16 + r + hi8) * N64 + tj * 16 + mloc] = f[r];
}

// ---------------------------------------------------------------------------
// Kernel 2: out[b] = F[ 12-bit pattern of bitstring b ]  (chip-wide gather)
// ---------------------------------------------------------------------------
__global__ __launch_bounds__(256)
void KOBE_gather_kernel(const int* __restrict__ bits,
                        const float* __restrict__ ws_F,
                        float* __restrict__ out) {
    int b = blockIdx.x * blockDim.x + threadIdx.x;
    if (b >= B_) return;
    const int* row = bits + b * NUMBITS_;
    int x = 0;
    #pragma unroll
    for (int i = 0; i < ORDER_; ++i)
        x |= (row[i] & 1) << i;        // spin i -> mask bit i
    out[b] = ws_F[x];
}

extern "C" void kernel_launch(void* const* d_in, const int* in_sizes, int n_in,
                              void* d_out, int out_size, void* d_ws, size_t ws_size,
                              hipStream_t stream) {
    const int*   bitstrings = (const int*)d_in[0];    // [4096, 32] int32
    const float* theta      = (const float*)d_in[1];  // [4095]     f32
    const int*   idx_pad    = (const int*)d_in[2];    // [4095, 12] int32
    float*       out        = (float*)d_out;          // [4096]     f32
    float*       ws_F       = (float*)d_ws;           // 4096 f32 scratch (16 KB)

    KOBE_wht_kernel<<<1, 512, 0, stream>>>(theta, idx_pad, ws_F);
    KOBE_gather_kernel<<<(B_ + 255) / 256, 256, 0, stream>>>(bitstrings, ws_F, out);
}
